// MultiHeadAttention_22943715295363
// MI455X (gfx1250) — matrix operand order repaired and checked by reading, not compiled, not run on hardware
//
#include <hip/hip_runtime.h>

typedef __attribute__((ext_vector_type(16))) __bf16 v16bf;
typedef __attribute__((ext_vector_type(8)))  __bf16 v8bf;
typedef __attribute__((ext_vector_type(8)))  float  v8f;
typedef __attribute__((ext_vector_type(4)))  unsigned int v4u;
typedef __attribute__((ext_vector_type(8)))  int v8i;
typedef __attribute__((ext_vector_type(4)))  int v4i;

#define LORA_SCALING 2.0f   // alpha/rank = 16/8
#define ATTN_SCALE   0.125f // 64^-0.5

// ---------------------------------------------------------------------------
// WMMA helper: D = A(16x32 bf16) x B(32x16 bf16) + C(16x16 f32)
// ---------------------------------------------------------------------------
__device__ __forceinline__ v8f wmma_bf16(v16bf a, v16bf b, v8f c) {
  return __builtin_amdgcn_wmma_f32_16x16x32_bf16(
      /*neg_a=*/false, a, /*neg_b=*/false, b,
      /*c_mod=*/(short)0, c, /*reuse_a=*/false, /*reuse_b=*/false);
}

union AFrag { v16bf v; v8bf h[2]; };

// A-matrix 16x32 bf16 fragment from row-major A[m][k] (contiguous k = 16*kh + s)
__device__ __forceinline__ v16bf load_a_frag(const __bf16* A, int lda,
                                             int m0, int k0, int lane) {
  const int m = lane & 15, kh = (lane >> 4) & 1;
  const __bf16* row = A + (size_t)(m0 + m) * lda + k0;
  AFrag f;
  f.h[0] = *(const v8bf*)(row + kh * 16);
  f.h[1] = *(const v8bf*)(row + kh * 16 + 8);
  return f.v;
}

// B-matrix 32x16 bf16 fragment where Bmat[k][n] = W[n][k] (row-major W)
__device__ __forceinline__ v16bf load_bT_frag(const __bf16* W, int ldw,
                                              int n0, int k0, int lane) {
  const int c = lane & 15, kh = (lane >> 4) & 1;
  return *(const v16bf*)(W + (size_t)(n0 + c) * ldw + k0 + kh * 16);
}

// ---------------------------------------------------------------------------
// TDM: stage a 2D tile (row-major) of bf16 from global into LDS.
// D# per CDNA5 ISA §8.3/8.4 (group0: count/lds/global/type, group1: dims).
// tile walks tile_d1 rows of tile_d0 elements, global row stride = stride0
// elements; LDS receives [tile_d1][tile_d0] row-major at lds_off.
// This toolchain exposes the 6-arg builtin (extra SGPR group, unused for 2D).
// ---------------------------------------------------------------------------
__device__ __forceinline__ void tdm_load_2d_bf16(unsigned lds_off,
                                                 const __bf16* gptr,
                                                 unsigned tile_d0, unsigned tile_d1,
                                                 unsigned tensor_d0, unsigned tensor_d1,
                                                 unsigned stride0) {
  const unsigned long long ga = (unsigned long long)(size_t)gptr;
  v4u g0;
  g0[0] = 1u;                                    // count=1 valid descriptor
  g0[1] = lds_off;                               // lds_addr (bytes)
  g0[2] = (unsigned)(ga & 0xffffffffu);          // global_addr[31:0]
  g0[3] = (unsigned)((ga >> 32) & 0x01ffffffu)   // global_addr[56:32]
        | (2u << 30);                            // type = 2 ("image")
  v8i g1;
  g1[0] = (int)(1u << 16);                       // wg_mask=0, data_size=1 (2B)
  g1[1] = (int)((tensor_d0 & 0xffffu) << 16);    // tensor_dim0[15:0]
  g1[2] = (int)((tensor_d0 >> 16) | ((tensor_d1 & 0xffffu) << 16));
  g1[3] = (int)((tensor_d1 >> 16) | (tile_d0 << 16));  // tile_dim0
  g1[4] = (int)(tile_d1 & 0xffffu);              // tile_dim1, tile_dim2=0
  g1[5] = (int)stride0;                          // tensor_dim0_stride[31:0]
  g1[6] = 0;                                     // stride0[47:32], stride1 lo
  g1[7] = 0;
  const v4i z4 = {0, 0, 0, 0};                   // groups 2/3 unused (2D)
  const v8i z8 = {0, 0, 0, 0, 0, 0, 0, 0};
  __builtin_amdgcn_tensor_load_to_lds(g0, g1, z4, z4, z8, 0);
}

// ---------------------------------------------------------------------------
// Elementwise f32 -> bf16 conversion
// ---------------------------------------------------------------------------
__global__ void cvt_f32_bf16(const float* __restrict__ in,
                             __bf16* __restrict__ out, int n) {
  int i = blockIdx.x * blockDim.x + threadIdx.x;
  const int stride = gridDim.x * blockDim.x;
  for (; i < n; i += stride) out[i] = (__bf16)in[i];
}

// ---------------------------------------------------------------------------
// LoRA first stage: xa[m][r] = LORA_SCALING * dot(x[m,:], A[r,:])  (rank 8)
// ---------------------------------------------------------------------------
__global__ void lora_xa_f32(const float* __restrict__ x,
                            const float* __restrict__ A,
                            float* __restrict__ xa, int M, int K) {
  const int idx = blockIdx.x * blockDim.x + threadIdx.x;
  if (idx >= M * 8) return;
  const int m = idx >> 3, r = idx & 7;
  const float* xr = x + (size_t)m * K;
  const float* ar = A + (size_t)r * K;
  float s = 0.f;
  for (int k = 0; k < K; ++k) s += xr[k] * ar[k];
  xa[idx] = s * LORA_SCALING;
}

__global__ void lora_xa_bf16(const __bf16* __restrict__ x,
                             const float* __restrict__ A,
                             float* __restrict__ xa, int M, int K) {
  const int idx = blockIdx.x * blockDim.x + threadIdx.x;
  if (idx >= M * 8) return;
  const int m = idx >> 3, r = idx & 7;
  const __bf16* xr = x + (size_t)m * K;
  const float* ar = A + (size_t)r * K;
  float s = 0.f;
  for (int k = 0; k < K; ++k) s += (float)xr[k] * ar[k];
  xa[idx] = s * LORA_SCALING;
}

// ---------------------------------------------------------------------------
// QKV GEMM: M=4096, N=3072, K=1024. Each wave: 32(M)x64(N) tile
// (2 A-frags reuse each B-frag -> half the B traffic per FLOP).
// Epilogue adds rank-8 LoRA, scatters bf16 into qbuf/kbuf [bh][s][d], vT [bh][d][s].
// ---------------------------------------------------------------------------
__global__ void __launch_bounds__(256)
qkv_gemm(const __bf16* __restrict__ xbf, const __bf16* __restrict__ wbf,
         const float* __restrict__ xa, const float* __restrict__ loraB,
         __bf16* __restrict__ qb, __bf16* __restrict__ kb,
         __bf16* __restrict__ vT) {
  const int lane = threadIdx.x & 31;
  const int wave = threadIdx.x >> 5;
  const int c = lane & 15, kh = (lane >> 4) & 1;
  const int n0 = blockIdx.x * 64;
  const int m0 = (blockIdx.y * 8 + wave) * 32;

  v8f acc[2][4] = {};
  for (int k0 = 0; k0 < 1024; k0 += 32) {
    v16bf a0 = load_a_frag(xbf, 1024, m0, k0, lane);
    v16bf a1 = load_a_frag(xbf, 1024, m0 + 16, k0, lane);
#pragma unroll
    for (int t = 0; t < 4; ++t) {
      v16bf b = load_bT_frag(wbf, 1024, n0 + 16 * t, k0, lane);
      acc[0][t] = wmma_bf16(a0, b, acc[0][t]);
      acc[1][t] = wmma_bf16(a1, b, acc[1][t]);
    }
  }

  float bn[4][8];
#pragma unroll
  for (int t = 0; t < 4; ++t) {
    const float* bp = loraB + (size_t)(n0 + 16 * t + c) * 8;
#pragma unroll
    for (int j = 0; j < 8; ++j) bn[t][j] = bp[j];
  }

  const int sel = n0 >> 10;            // 0=Q 1=K 2=V
  const int h = (n0 & 1023) >> 6;      // head

#pragma unroll
  for (int half = 0; half < 2; ++half) {
    const int mb = m0 + 16 * half;
    const int batch = mb >> 11;
    const int bh = batch * 16 + h;
    const int sbase = (mb & 2047) + kh * 8;
#pragma unroll
    for (int r = 0; r < 8; ++r) {
      const int mrow = mb + r + kh * 8;
      const float* xrow = xa + (size_t)mrow * 8;
      float xr[8];
#pragma unroll
      for (int j = 0; j < 8; ++j) xr[j] = xrow[j];
      const int s = sbase + r;
#pragma unroll
      for (int t = 0; t < 4; ++t) {
        float v = acc[half][t][r];
#pragma unroll
        for (int j = 0; j < 8; ++j) v += xr[j] * bn[t][j];
        const int d = 16 * t + c;
        const __bf16 bv = (__bf16)v;
        if (sel == 0)      qb[((size_t)bh * 2048 + s) * 64 + d] = bv;
        else if (sel == 1) kb[((size_t)bh * 2048 + s) * 64 + d] = bv;
        else               vT[((size_t)bh * 64 + d) * 2048 + s] = bv;
      }
    }
  }
}

// ---------------------------------------------------------------------------
// Flash attention. One block = 8 waves sharing one (b,h); each wave owns a
// 16-row Q tile. K/V 32-key tiles are staged into double-buffered LDS by the
// Tensor Data Mover (issued by wave 0, overlapped with compute of the current
// tile), published via s_wait_tensorcnt + workgroup barrier. This removes the
// 8x redundant per-wave K/V global streams.
// ---------------------------------------------------------------------------
__global__ void __launch_bounds__(256)
attn_kernel(const __bf16* __restrict__ qb, const __bf16* __restrict__ kbuf,
            const __bf16* __restrict__ vT, const int* __restrict__ mask,
            __bf16* __restrict__ attn_out) {
  __shared__ __align__(64) __bf16 kstage[2][32 * 64];  // [key][d]
  __shared__ __align__(64) __bf16 vstage[2][64 * 32];  // [d][s]
  __shared__ __align__(64) __bf16 pstage[8][512];      // per-wave 16x32 P

  const int lane = threadIdx.x & 31;
  const int wave = threadIdx.x >> 5;
  const int c = lane & 15, kh = (lane >> 4) & 1;
  const int bh = blockIdx.x;
  const int q0 = (blockIdx.y * 8 + wave) * 16;
  const __bf16* qbase = qb + (size_t)bh * 2048 * 64;
  const __bf16* kbase = kbuf + (size_t)bh * 2048 * 64;
  const __bf16* vbase = vT + (size_t)bh * 64 * 2048;
  __bf16* pw = &pstage[wave][0];

  const unsigned kst_off[2] = {(unsigned)(size_t)(const void*)&kstage[0][0],
                               (unsigned)(size_t)(const void*)&kstage[1][0]};
  const unsigned vst_off[2] = {(unsigned)(size_t)(const void*)&vstage[0][0],
                               (unsigned)(size_t)(const void*)&vstage[1][0]};

  const v16bf aq0 = load_a_frag(qbase, 64, q0, 0, lane);
  const v16bf aq1 = load_a_frag(qbase, 64, q0, 32, lane);

  v8f accO[4] = {};
  float mrow[8], lrow[8];
#pragma unroll
  for (int r = 0; r < 8; ++r) { mrow[r] = -3.0e38f; lrow[r] = 0.f; }

  // Prologue: DMA first K/V tile into buffer 0.
  if (wave == 0) {
    tdm_load_2d_bf16(kst_off[0], kbase, 64, 32, 64, 2048, 64);
    tdm_load_2d_bf16(vst_off[0], vbase, 32, 64, 2048, 64, 2048);
  }

  for (int i = 0; i < 64; ++i) {
    const int kb = i * 32;
    const int cur = i & 1, nxt = cur ^ 1;

    if (wave == 0) __builtin_amdgcn_s_wait_tensorcnt(0);
    __syncthreads();  // current tile visible; previous-buffer readers done

    if (wave == 0 && i + 1 < 64) {  // overlap next DMA with this tile's math
      tdm_load_2d_bf16(kst_off[nxt], kbase + (size_t)(kb + 32) * 64,
                       64, 32, 64, 2048, 64);
      tdm_load_2d_bf16(vst_off[nxt], vbase + (kb + 32),
                       32, 64, 2048, 64, 2048);
    }

    const __bf16* ks = &kstage[cur][0];
    const __bf16* vs = &vstage[cur][0];

    v8f s0 = {}, s1 = {};
    s0 = wmma_bf16(aq0, load_bT_frag(ks, 64, 0, 0, lane), s0);
    s0 = wmma_bf16(aq1, load_bT_frag(ks, 64, 0, 32, lane), s0);
    s1 = wmma_bf16(aq0, load_bT_frag(ks, 64, 16, 0, lane), s1);
    s1 = wmma_bf16(aq1, load_bT_frag(ks, 64, 16, 32, lane), s1);

#pragma unroll
    for (int r = 0; r < 8; ++r) {
      const int qrow = q0 + r + kh * 8;
      const int* mp = mask + (size_t)qrow * 2048 + kb;
      float v0 = mp[c]      ? s0[r] * ATTN_SCALE : -1.0e9f;
      float v1 = mp[16 + c] ? s1[r] * ATTN_SCALE : -1.0e9f;
      float rm = fmaxf(v0, v1);
#pragma unroll
      for (int off = 8; off >= 1; off >>= 1) rm = fmaxf(rm, __shfl_xor(rm, off));
      const float mnew = fmaxf(mrow[r], rm);
      const float corr = __expf(mrow[r] - mnew);
      const float p0 = __expf(v0 - mnew);
      const float p1 = __expf(v1 - mnew);
      float rs = p0 + p1;
#pragma unroll
      for (int off = 8; off >= 1; off >>= 1) rs += __shfl_xor(rs, off);
      lrow[r] = lrow[r] * corr + rs;
      mrow[r] = mnew;
#pragma unroll
      for (int t = 0; t < 4; ++t) accO[t][r] *= corr;
      const int prow = r + kh * 8;
      pw[prow * 32 + c]      = (__bf16)p0;
      pw[prow * 32 + 16 + c] = (__bf16)p1;
    }
    __builtin_amdgcn_wave_barrier();  // order per-wave LDS store -> load
    const v16bf ap = load_a_frag(pw, 32, 0, 0, lane);
    __builtin_amdgcn_wave_barrier();
#pragma unroll
    for (int t = 0; t < 4; ++t) {
      v16bf bv = load_bT_frag(vs, 32, 16 * t, 0, lane);
      accO[t] = wmma_bf16(ap, bv, accO[t]);
    }
  }

  const int batch = bh >> 4, h = bh & 15;
#pragma unroll
  for (int r = 0; r < 8; ++r) {
    const int s = q0 + r + kh * 8;
    const float inv = 1.0f / lrow[r];
#pragma unroll
    for (int t = 0; t < 4; ++t) {
      attn_out[((size_t)(batch * 2048 + s)) * 1024 + h * 64 + 16 * t + c] =
          (__bf16)(accO[t][r] * inv);
    }
  }
}

// ---------------------------------------------------------------------------
// Output projection: out = attn @ W.T + bias + oa @ loraB.T   (f32 out)
// Each wave: 32(M)x64(N) tile.
// ---------------------------------------------------------------------------
__global__ void __launch_bounds__(256)
proj_gemm(const __bf16* __restrict__ abf, const __bf16* __restrict__ wbf,
          const float* __restrict__ oa, const float* __restrict__ loraB,
          const float* __restrict__ bias, float* __restrict__ out) {
  const int lane = threadIdx.x & 31;
  const int wave = threadIdx.x >> 5;
  const int c = lane & 15, kh = (lane >> 4) & 1;
  const int n0 = blockIdx.x * 64;
  const int m0 = (blockIdx.y * 8 + wave) * 32;

  v8f acc[2][4] = {};
  for (int k0 = 0; k0 < 1024; k0 += 32) {
    v16bf a0 = load_a_frag(abf, 1024, m0, k0, lane);
    v16bf a1 = load_a_frag(abf, 1024, m0 + 16, k0, lane);
#pragma unroll
    for (int t = 0; t < 4; ++t) {
      v16bf b = load_bT_frag(wbf, 1024, n0 + 16 * t, k0, lane);
      acc[0][t] = wmma_bf16(a0, b, acc[0][t]);
      acc[1][t] = wmma_bf16(a1, b, acc[1][t]);
    }
  }

  float bn[4][8], bb[4];
#pragma unroll
  for (int t = 0; t < 4; ++t) {
    const int n = n0 + 16 * t + c;
    bb[t] = bias[n];
    const float* bp = loraB + (size_t)n * 8;
#pragma unroll
    for (int j = 0; j < 8; ++j) bn[t][j] = bp[j];
  }

#pragma unroll
  for (int half = 0; half < 2; ++half) {
    const int mb = m0 + 16 * half;
#pragma unroll
    for (int r = 0; r < 8; ++r) {
      const int mrow = mb + r + kh * 8;
      const float* orow = oa + (size_t)mrow * 8;
      float xr[8];
#pragma unroll
      for (int j = 0; j < 8; ++j) xr[j] = orow[j];
#pragma unroll
      for (int t = 0; t < 4; ++t) {
        float v = acc[half][t][r] + bb[t];
#pragma unroll
        for (int j = 0; j < 8; ++j) v += xr[j] * bn[t][j];
        out[(size_t)mrow * 1024 + n0 + 16 * t + c] = v;
      }
    }
  }
}

// ---------------------------------------------------------------------------
extern "C" void kernel_launch(void* const* d_in, const int* in_sizes, int n_in,
                              void* d_out, int out_size, void* d_ws, size_t ws_size,
                              hipStream_t stream) {
  const float* x      = (const float*)d_in[0];
  const int*   mask   = (const int*)d_in[1];
  const float* qkv_w  = (const float*)d_in[2];
  const float* proj_w = (const float*)d_in[3];
  const float* proj_b = (const float*)d_in[4];
  const float* lAq    = (const float*)d_in[5];
  const float* lBq    = (const float*)d_in[6];
  const float* lAp    = (const float*)d_in[7];
  const float* lBp    = (const float*)d_in[8];
  float* out = (float*)d_out;

  char* ws = (char*)d_ws;
  __bf16* xbf   = (__bf16*)(ws + 0);         // 8 MiB
  __bf16* wqkv  = (__bf16*)(ws + 8388608);   // 6 MiB
  __bf16* wproj = (__bf16*)(ws + 14680064);  // 2 MiB
  __bf16* qbuf  = (__bf16*)(ws + 16777216);  // 8 MiB
  __bf16* kbuf  = (__bf16*)(ws + 25165824);  // 8 MiB
  __bf16* vT    = (__bf16*)(ws + 33554432);  // 8 MiB (d-major)
  __bf16* attnb = (__bf16*)(ws + 41943040);  // 8 MiB
  float*  xa    = (float*)(ws + 50331648);   // 128 KiB
  float*  oa    = (float*)(ws + 50462720);   // 128 KiB

  cvt_f32_bf16<<<1024, 256, 0, stream>>>(x, xbf, 4096 * 1024);
  cvt_f32_bf16<<<1024, 256, 0, stream>>>(qkv_w, wqkv, 3072 * 1024);
  cvt_f32_bf16<<<512, 256, 0, stream>>>(proj_w, wproj, 1024 * 1024);
  lora_xa_f32<<<128, 256, 0, stream>>>(x, lAq, xa, 4096, 1024);
  qkv_gemm<<<dim3(48, 16), 256, 0, stream>>>(xbf, wqkv, xa, lBq, qbuf, kbuf, vT);
  attn_kernel<<<dim3(32, 16), 256, 0, stream>>>(qbuf, kbuf, vT, mask, attnb);
  lora_xa_bf16<<<128, 256, 0, stream>>>(attnb, lAp, oa, 4096, 1024);
  proj_gemm<<<dim3(16, 16), 256, 0, stream>>>(attnb, wproj, oa, lBp, proj_b, out);
}
